// MultiHeadAttention_88218628260446
// MI455X (gfx1250) — matrix operand order repaired and checked by reading, not compiled, not run on hardware
//
#include <hip/hip_runtime.h>
#include <hip/hip_bf16.h>

#define EMB 1024
#define SEQ 2048
#define NH  16
#define HD  64
#define NB  4

typedef _Float16 h16;
typedef __attribute__((ext_vector_type(16))) _Float16 v16h;
typedef __attribute__((ext_vector_type(8)))  _Float16 v8h;
typedef __attribute__((ext_vector_type(8)))  float    v8f;

// log2(e) / sqrt(HD): softmax done in base-2 domain (exact up to fp assoc.)
#define SCALE2 0.1803368801111204f

__device__ __forceinline__ v8f wmma_f16(v16h a, v16h b, v8f c) {
  return __builtin_amdgcn_wmma_f32_16x16x32_f16(
      /*neg_a=*/false, a, /*neg_b=*/false, b,
      /*c_mod=*/(short)0, c, /*reuse_a=*/false, /*reuse_b=*/false);
}

// A-fragment (16x32, f16 row-major src). ISA layout: lane<16 holds row=lane,
// K={0..7,16..23}; lane>=16 holds row=lane-16, K={8..15,24..31}.
__device__ __forceinline__ v16h frag_a_h(const h16* __restrict__ base, int lda) {
  const int lane = threadIdx.x & 31;
  const int row  = lane & 15;
  const int kb   = (lane >> 4) << 3;         // 0 or 8
  const h16* p = base + (size_t)row * lda + kb;
  v8h lo = *(const v8h*)(p);
  v8h hi = *(const v8h*)(p + 16);
  return __builtin_shufflevector(lo, hi, 0,1,2,3,4,5,6,7,8,9,10,11,12,13,14,15);
}

// Same, converting from f32 row-major source.
__device__ __forceinline__ v16h frag_a_f(const float* __restrict__ base, int lda) {
  const int lane = threadIdx.x & 31;
  const int row  = lane & 15;
  const int kb   = (lane >> 4) << 3;
  const float* p = base + (size_t)row * lda + kb;
  v16h a;
#pragma unroll
  for (int j = 0; j < 8; ++j) { a[j] = (h16)p[j]; a[8 + j] = (h16)p[16 + j]; }
  return a;
}

// B-fragment (32x16): B[k][n] = src[n*ldb + k] (i.e. B = srcT, src row-major).
// Lane holds col n=lane&15, K = {kb..kb+7, kb+16..kb+23} with kb = (lane>=16)*8.
__device__ __forceinline__ v16h frag_b_h(const h16* __restrict__ base, int ldb) {
  const int lane = threadIdx.x & 31;
  const int n  = lane & 15;
  const int kb = (lane >> 4) << 3;           // 0 or 8
  const h16* p = base + (size_t)n * ldb + kb;
  v8h lo = *(const v8h*)(p);
  v8h hi = *(const v8h*)(p + 16);
  return __builtin_shufflevector(lo, hi, 0,1,2,3,4,5,6,7,8,9,10,11,12,13,14,15);
}

__device__ __forceinline__ v16h frag_b_f(const float* __restrict__ base, int ldb) {
  const int lane = threadIdx.x & 31;
  const int n  = lane & 15;
  const int kb = (lane >> 4) << 3;
  const float* p = base + (size_t)n * ldb + kb;
  v16h b;
#pragma unroll
  for (int j = 0; j < 8; ++j) { b[j] = (h16)p[j]; b[8 + j] = (h16)p[16 + j]; }
  return b;
}

// 16-lane (half-wave) reductions: C-fragment rows live inside one half-wave.
__device__ __forceinline__ float hmax16(float v) {
#pragma unroll
  for (int m = 1; m < 16; m <<= 1) v = fmaxf(v, __shfl_xor(v, m, 32));
  return v;
}
__device__ __forceinline__ float hsum16(float v) {
#pragma unroll
  for (int m = 1; m < 16; m <<= 1) v += __shfl_xor(v, m, 32);
  return v;
}

// Async copy of one 16x64 f16 K-tile (2 KB) into LDS: 128 threads x 16 B.
// Tracked by ASYNCcnt; drained with s_wait_asynccnt before a block barrier.
__device__ __forceinline__ void stage_k_async(const h16* __restrict__ gbase,
                                              h16* lbase) {
  const int t = threadIdx.x;                       // 0..127
  const h16* g = gbase + t * 8;                    // 16 bytes per thread
  unsigned lofs = (unsigned)(unsigned long long)(lbase + t * 8);  // LDS offset
  asm volatile("global_load_async_to_lds_b128 %0, %1, off"
               :: "v"(lofs), "v"(g) : "memory");
}

// ---------------- 1) QKV projection: Y = X @ W^T + b, f32 -> f16 -------------
// mode 0: Y[((b*NH+h)*SEQ+s)*HD+d]   (Q, K head layout)
// mode 1: Y[((b*NH+h)*HD+d)*SEQ+s]   (V transposed for contiguous PV B-frags)
__global__ __launch_bounds__(128) void proj_qkv_kernel(
    const float* __restrict__ X, const float* __restrict__ W,
    const float* __restrict__ bias, h16* __restrict__ Y, int mode) {
  const int wave = threadIdx.x >> 5;
  const int lane = threadIdx.x & 31;
  const int m0 = blockIdx.x * 64 + (wave & 1) * 32;
  const int n0 = blockIdx.y * 64 + (wave >> 1) * 32;
  v8f acc[2][2] = {{{}, {}}, {{}, {}}};
  for (int k0 = 0; k0 < EMB; k0 += 32) {
    v16h a0 = frag_a_f(X + (size_t)m0 * EMB + k0, EMB);
    v16h a1 = frag_a_f(X + (size_t)(m0 + 16) * EMB + k0, EMB);
    v16h b0 = frag_b_f(W + (size_t)n0 * EMB + k0, EMB);
    v16h b1 = frag_b_f(W + (size_t)(n0 + 16) * EMB + k0, EMB);
    acc[0][0] = wmma_f16(a0, b0, acc[0][0]);
    acc[0][1] = wmma_f16(a0, b1, acc[0][1]);
    acc[1][0] = wmma_f16(a1, b0, acc[1][0]);
    acc[1][1] = wmma_f16(a1, b1, acc[1][1]);
  }
  const int hi = lane >> 4, nn = lane & 15;
  float bj[2];
#pragma unroll
  for (int j = 0; j < 2; ++j) bj[j] = bias[n0 + j * 16 + nn];
#pragma unroll
  for (int i = 0; i < 2; ++i)
#pragma unroll
    for (int j = 0; j < 2; ++j)
#pragma unroll
      for (int r = 0; r < 8; ++r) {
        int m = m0 + i * 16 + r + 8 * hi;
        int n = n0 + j * 16 + nn;
        float v = acc[i][j][r] + bj[j];
        int bb = m >> 11, s = m & (SEQ - 1);
        int h = n >> 6,  d = n & (HD - 1);
        size_t idx = (mode == 0)
            ? ((((size_t)bb * NH + h) * SEQ + s) * HD + d)
            : ((((size_t)bb * NH + h) * HD + d) * SEQ + s);
        Y[idx] = (h16)v;
      }
}

// ---------------- 2) softmax stats (row max + sum-exp2) ----------------------
__global__ __launch_bounds__(128) void attn_stats_kernel(
    const h16* __restrict__ Qh, const h16* __restrict__ Kh,
    float* __restrict__ mW, float* __restrict__ lW) {
  const int wave = threadIdx.x >> 5;
  const int lane = threadIdx.x & 31;
  const int b = blockIdx.z, h = blockIdx.y;
  const int q0 = blockIdx.x * 64 + wave * 16;
  const size_t head = (size_t)(b * NH + h);
  const h16* qb = Qh + (head * SEQ + q0) * HD;
  const h16* kb0 = Kh + head * SEQ * HD;
  v16h aQ0 = frag_a_h(qb, HD);
  v16h aQ1 = frag_a_h(qb + 32, HD);
  float m2[8], lsum[8];
#pragma unroll
  for (int r = 0; r < 8; ++r) { m2[r] = -3.0e38f; lsum[r] = 0.f; }
  for (int kv = 0; kv < SEQ; kv += 16) {
    const h16* kb = kb0 + (size_t)kv * HD;
    __builtin_prefetch(kb + 16 * HD, 0, 1);   // global_prefetch_b8 on next tile
    v16h bK0 = frag_b_h(kb, HD);
    v16h bK1 = frag_b_h(kb + 32, HD);
    v8f c = {};
    c = wmma_f16(aQ0, bK0, c);
    c = wmma_f16(aQ1, bK1, c);
#pragma unroll
    for (int r = 0; r < 8; ++r) {
      float s2 = c[r] * SCALE2;
      float mnew = fmaxf(m2[r], hmax16(s2));
      float e = hsum16(exp2f(s2 - mnew));
      lsum[r] = lsum[r] * exp2f(m2[r] - mnew) + e;
      m2[r] = mnew;
    }
  }
  if ((lane & 15) == 0) {
    const int hi = lane >> 4;
#pragma unroll
    for (int r = 0; r < 8; ++r) {
      size_t ridx = head * SEQ + q0 + r + 8 * hi;
      mW[ridx] = m2[r];
      lW[ridx] = lsum[r];
    }
  }
}

// ---------------- 3) O = softmax(QK^T) @ V  (P relayout via LDS) -------------
__global__ __launch_bounds__(128) void attn_out_kernel(
    const h16* __restrict__ Qh, const h16* __restrict__ Kh,
    const h16* __restrict__ Vt, const float* __restrict__ mW,
    const float* __restrict__ lW, h16* __restrict__ AO) {
  __shared__ h16 Plds[4][16 * 32];   // per-wave 16x32 P tile, 1 KB each
  const int wave = threadIdx.x >> 5;
  const int lane = threadIdx.x & 31;
  const int hi = lane >> 4, nn = lane & 15;
  const int b = blockIdx.z, h = blockIdx.y;
  const int q0 = blockIdx.x * 64 + wave * 16;
  const size_t head = (size_t)(b * NH + h);
  const h16* qb = Qh + (head * SEQ + q0) * HD;
  const h16* kb0 = Kh + head * SEQ * HD;
  const h16* vb0 = Vt + head * (size_t)HD * SEQ;
  v16h aQ0 = frag_a_h(qb, HD);
  v16h aQ1 = frag_a_h(qb + 32, HD);
  float m2[8], rl[8];
#pragma unroll
  for (int r = 0; r < 8; ++r) {
    size_t ridx = head * SEQ + q0 + r + 8 * hi;
    m2[r] = mW[ridx];
    rl[r] = 1.0f / lW[ridx];
  }
  h16* P = &Plds[wave][0];
  v8f o[4] = {{}, {}, {}, {}};
  for (int kv = 0; kv < SEQ; kv += 32) {
#pragma unroll
    for (int cc = 0; cc < 2; ++cc) {
      const h16* kb = kb0 + (size_t)(kv + cc * 16) * HD;
      __builtin_prefetch(kb + 32 * HD, 0, 1);
      v16h bK0 = frag_b_h(kb, HD);
      v16h bK1 = frag_b_h(kb + 32, HD);
      v8f c = {};
      c = wmma_f16(aQ0, bK0, c);
      c = wmma_f16(aQ1, bK1, c);
#pragma unroll
      for (int r = 0; r < 8; ++r) {
        float p = exp2f(c[r] * SCALE2 - m2[r]);   // in (0,1]: f16-safe
        P[(r + 8 * hi) * 32 + cc * 16 + nn] = (h16)p;
      }
    }
    // wave-private LDS round-trip (C-layout -> A-fragment); LDS is in-order
    // per wave, just keep the compiler from reordering and drain DScnt.
    __builtin_amdgcn_wave_barrier();
    asm volatile("s_wait_dscnt 0" ::: "memory");
    v16h aP = frag_a_h(P, 32);
    __builtin_amdgcn_wave_barrier();
#pragma unroll
    for (int t = 0; t < 4; ++t) {
      v16h bV = frag_b_h(vb0 + (size_t)(t * 16) * SEQ + kv, SEQ);
      o[t] = wmma_f16(aP, bV, o[t]);
    }
  }
#pragma unroll
  for (int t = 0; t < 4; ++t)
#pragma unroll
    for (int r = 0; r < 8; ++r) {
      int row = q0 + r + 8 * hi;
      float v = o[t][r] * rl[r];
      AO[((size_t)b * SEQ + row) * EMB + h * HD + t * 16 + nn] = (h16)v;
    }
}

// ---------------- 4) attn.mean(heads): recompute P, sum over h ---------------
// All 4 waves consume the SAME K tile per (head, kv) chunk: stage it once per
// block with GLOBAL_LOAD_ASYNC_TO_LDS_B128 (double-buffered over the head
// loop, ASYNCcnt-tracked) instead of 4x redundant vector loads.
__global__ __launch_bounds__(128) void attn_mean_kernel(
    const h16* __restrict__ Qh, const h16* __restrict__ Kh,
    const float* __restrict__ mW, const float* __restrict__ lW,
    float* __restrict__ mean) {
  __shared__ float mS[4][NH][16];
  __shared__ float rS[4][NH][16];
  __shared__ h16 Kbuf[2][16 * HD];          // double-buffered 16x64 K tile
  const int wave = threadIdx.x >> 5;
  const int lane = threadIdx.x & 31;
  const int b = blockIdx.y;
  const int q0 = blockIdx.x * 64 + wave * 16;
  for (int idx = lane; idx < NH * 16; idx += 32) {
    int hh = idx >> 4, rr = idx & 15;
    size_t ridx = ((size_t)(b * NH + hh)) * SEQ + q0 + rr;
    mS[wave][hh][rr] = mW[ridx];
    rS[wave][hh][rr] = 1.0f / lW[ridx];
  }
  __builtin_amdgcn_wave_barrier();
  asm volatile("s_wait_dscnt 0" ::: "memory");
  const int hi = lane >> 4, nn = lane & 15;
  const size_t head0 = (size_t)b * NH;
  for (int kv = 0; kv < SEQ; kv += 16) {
    float macc[8] = {0.f, 0.f, 0.f, 0.f, 0.f, 0.f, 0.f, 0.f};
    // prologue: stage heads 0 and 1
    stage_k_async(Kh + ((head0 + 0) * SEQ + kv) * HD, &Kbuf[0][0]);
    stage_k_async(Kh + ((head0 + 1) * SEQ + kv) * HD, &Kbuf[1][0]);
    for (int hh = 0; hh < NH; ++hh) {
      const int cur = hh & 1;
      // async loads complete in order: <=1 outstanding => current tile landed
      asm volatile("s_wait_asynccnt 0x1" ::: "memory");
      __syncthreads();                       // publish all waves' portions
      const h16* qb = Qh + ((head0 + hh) * SEQ + q0) * HD;
      v16h a0 = frag_a_h(qb, HD);
      v16h a1 = frag_a_h(qb + 32, HD);
      v16h b0 = frag_b_h(&Kbuf[cur][0], HD);
      v16h b1 = frag_b_h(&Kbuf[cur][32], HD);
      v8f c = {};
      c = wmma_f16(a0, b0, c);
      c = wmma_f16(a1, b1, c);
#pragma unroll
      for (int r = 0; r < 8; ++r) {
        int row = r + 8 * hi;
        macc[r] += exp2f(c[r] * SCALE2 - mS[wave][hh][row]) * rS[wave][hh][row];
      }
      __syncthreads();                       // buffer free before overwrite
      if (hh + 2 < NH)
        stage_k_async(Kh + ((head0 + hh + 2) * SEQ + kv) * HD, &Kbuf[cur][0]);
    }
#pragma unroll
    for (int r = 0; r < 8; ++r) {
      int row = q0 + r + 8 * hi;
      mean[((size_t)b * SEQ + row) * SEQ + kv + nn] = macc[r] * (1.0f / NH);
    }
  }
}

// ---------------- 5) output projection: out = AO @ Wo^T + bo (f32 out) -------
__global__ __launch_bounds__(128) void proj_o_kernel(
    const h16* __restrict__ AO, const float* __restrict__ Wo,
    const float* __restrict__ bo, float* __restrict__ out) {
  const int wave = threadIdx.x >> 5;
  const int lane = threadIdx.x & 31;
  const int m0 = blockIdx.x * 64 + (wave & 1) * 32;
  const int n0 = blockIdx.y * 64 + (wave >> 1) * 32;
  v8f acc[2][2] = {{{}, {}}, {{}, {}}};
  for (int k0 = 0; k0 < EMB; k0 += 32) {
    v16h a0 = frag_a_h(AO + (size_t)m0 * EMB + k0, EMB);
    v16h a1 = frag_a_h(AO + (size_t)(m0 + 16) * EMB + k0, EMB);
    v16h b0 = frag_b_f(Wo + (size_t)n0 * EMB + k0, EMB);
    v16h b1 = frag_b_f(Wo + (size_t)(n0 + 16) * EMB + k0, EMB);
    acc[0][0] = wmma_f16(a0, b0, acc[0][0]);
    acc[0][1] = wmma_f16(a0, b1, acc[0][1]);
    acc[1][0] = wmma_f16(a1, b0, acc[1][0]);
    acc[1][1] = wmma_f16(a1, b1, acc[1][1]);
  }
  const int hi = lane >> 4, nn = lane & 15;
  float bj[2];
#pragma unroll
  for (int j = 0; j < 2; ++j) bj[j] = bo[n0 + j * 16 + nn];
#pragma unroll
  for (int i = 0; i < 2; ++i)
#pragma unroll
    for (int j = 0; j < 2; ++j)
#pragma unroll
      for (int r = 0; r < 8; ++r) {
        int m = m0 + i * 16 + r + 8 * hi;
        int n = n0 + j * 16 + nn;
        out[(size_t)m * EMB + n] = acc[i][j][r] + bj[j];
      }
}

extern "C" void kernel_launch(void* const* d_in, const int* in_sizes, int n_in,
                              void* d_out, int out_size, void* d_ws, size_t ws_size,
                              hipStream_t stream) {
  const float* query = (const float*)d_in[0];
  const float* key_  = (const float*)d_in[1];
  const float* value = (const float*)d_in[2];
  const float* Wq = (const float*)d_in[3];
  const float* bq = (const float*)d_in[4];
  const float* Wk = (const float*)d_in[5];
  const float* bk = (const float*)d_in[6];
  const float* Wv = (const float*)d_in[7];
  const float* bv = (const float*)d_in[8];
  const float* Wo = (const float*)d_in[9];
  const float* bo = (const float*)d_in[10];

  // Workspace layout (~65 MB):
  const size_t nQKV = (size_t)NB * NH * SEQ * HD;   // 8,388,608
  h16* Qh = (h16*)d_ws;
  h16* Kh = Qh + nQKV;
  h16* Vt = Kh + nQKV;
  h16* AO = Vt + nQKV;                              // [B, S, E] f16
  float* mW = (float*)(AO + (size_t)NB * SEQ * EMB);
  float* lW = mW + (size_t)NB * NH * SEQ;

  float* out_main = (float*)d_out;                           // [B,S,E]
  float* out_mean = out_main + (size_t)NB * SEQ * EMB;       // [B,S,S]

  dim3 blk(128);
  dim3 gProj(SEQ * NB / 64, EMB / 64);          // (128, 16)
  proj_qkv_kernel<<<gProj, blk, 0, stream>>>(query, Wq, bq, Qh, 0);
  proj_qkv_kernel<<<gProj, blk, 0, stream>>>(key_,  Wk, bk, Kh, 0);
  proj_qkv_kernel<<<gProj, blk, 0, stream>>>(value, Wv, bv, Vt, 1);

  dim3 gAttn(SEQ / 64, NH, NB);                 // (32, 16, 4)
  attn_stats_kernel<<<gAttn, blk, 0, stream>>>(Qh, Kh, mW, lW);
  attn_out_kernel<<<gAttn, blk, 0, stream>>>(Qh, Kh, Vt, mW, lW, AO);

  dim3 gMean(SEQ / 64, NB);                     // (32, 4)
  attn_mean_kernel<<<gMean, blk, 0, stream>>>(Qh, Kh, mW, lW, out_mean);

  proj_o_kernel<<<gProj, blk, 0, stream>>>(AO, Wo, bo, out_main);
}
